// E_GCL_2791728742861
// MI455X (gfx1250) — matrix operand order repaired and checked by reading, not compiled, not run on hardware
//
#include <hip/hip_runtime.h>

#define NN 20000   // nodes
#define NE 20000   // edges (E == N required by reference)
#define INF 64
#define HID 64
#define OUTF 67

typedef __attribute__((ext_vector_type(16))) _Float16 v16h;
typedef __attribute__((ext_vector_type(8)))  _Float16 v8h;
typedef __attribute__((ext_vector_type(8)))  float    v8f;

// ---------------- helpers ----------------
__device__ __forceinline__ float siluf(float x) { return x / (1.0f + __expf(-x)); }

struct U16B { unsigned int w[4]; };
__device__ __forceinline__ U16B ldu(const void* p) { U16B r; __builtin_memcpy(&r, p, 16); return r; }
__device__ __forceinline__ void stu(void* p, const U16B& v) { __builtin_memcpy(p, &v, 16); }
__device__ __forceinline__ v8h  ld8h(const void* p)  { v8h r;  __builtin_memcpy(&r, p, 16); return r; }
__device__ __forceinline__ v16h ld16h(const void* p) { v16h r; __builtin_memcpy(&r, p, 32); return r; }

#define WMMA16(A, B, C) \
  __builtin_amdgcn_wmma_f32_16x16x32_f16(false, (A), false, (B), (short)0, (C), false, false)

// ---------------- workspace layout ----------------
static constexpr size_t AL(size_t x) { return (x + 255) & ~size_t(255); }
static constexpr size_t O_EH  = 0;                                    // e (f16)  [E,64]
static constexpr size_t O_CD  = AL(O_EH  + (size_t)NE * 64 * 2);      // coord_diff f32 [E,3]
static constexpr size_t O_REL = AL(O_CD  + (size_t)NE * 3 * 4);       // rel f32 [E,3]
static constexpr size_t O_EF  = AL(O_REL + (size_t)NE * 3 * 4);       // edge_feat f32 [E,64]
static constexpr size_t O_W1F = AL(O_EF  + (size_t)NE * 64 * 4);      // Ws1 frags f16 [4096*128]
static constexpr size_t O_W2F = AL(O_W1F + (size_t)4096 * 128 * 2);   // Ws2 frags f16 [128*64]
static constexpr size_t O_NUM = AL(O_W2F + (size_t)128 * 64 * 2);     // num f32 [N,3]
static constexpr size_t O_CNT = AL(O_NUM + (size_t)NN * 3 * 4);       // cnt f32 [N]
static constexpr size_t O_AGG = AL(O_CNT + (size_t)NN * 4);           // agg f32 [N,64]

// ---------------- kernel 1: convert Ws1/Ws2 to WMMA B-fragment layout ----------------
// B fragment (f16, 32x16) as built here (A uses same k order): lane L holds column N = L%16; half h holds K_local = (L/16)*16 + h.
// Frag index f: Ws1 -> f = kstep*8 + ntile (kstep 0..127, ntile 0..7); Ws2 -> f = kstep*4 + ntile.
__global__ void prep_w_kernel(const float* __restrict__ Ws1, const float* __restrict__ Ws2,
                              _Float16* __restrict__ w1f, _Float16* __restrict__ w2f) {
  const int tid = blockIdx.x * 256 + threadIdx.x;
  const int total1 = 4096 * 128;
  if (tid < total1) {
    const int f = tid >> 9, rsub = tid & 511;
    const int lane = rsub >> 4, hh = rsub & 15;
    const int kstep = f >> 3, nt = f & 7;
    const int row = kstep * 32 + (lane >> 4) * 16 + hh;   // K in [0,4096)
    const int col = nt * 16 + (lane & 15);                // N in [0,128)
    w1f[tid] = (_Float16)Ws1[row * 128 + col];
  } else if (tid < total1 + 128 * 64) {
    const int t = tid - total1;
    const int f = t >> 9, rsub = t & 511;
    const int lane = rsub >> 4, hh = rsub & 15;
    const int kstep = f >> 2, nt = f & 3;
    const int row = kstep * 32 + (lane >> 4) * 16 + hh;   // K in [0,128)
    const int col = nt * 16 + (lane & 15);                // N in [0,64)
    w2f[t] = (_Float16)Ws2[row * 64 + col];
  }
}

// ---------------- kernel 2: zero accumulators ----------------
__global__ void zero_kernel(float* __restrict__ p, int n) {
  const int i = blockIdx.x * 256 + threadIdx.x;
  if (i < n) p[i] = 0.0f;
}

// ---------------- kernel 3: geometry + edge MLP ----------------
__global__ __launch_bounds__(128) void edge_mlp_kernel(
    const float* __restrict__ h, const float* __restrict__ coord, const int* __restrict__ ei,
    const float* __restrict__ We1, const float* __restrict__ be1,
    const float* __restrict__ We2, const float* __restrict__ be2,
    _Float16* __restrict__ eH, float* __restrict__ cd, float* __restrict__ rel) {
  __shared__ __align__(16) float W1[129 * 64];
  __shared__ __align__(16) float W2[64 * 64];
  __shared__ float B1[64];
  __shared__ float B2[64];
  for (int i = threadIdx.x; i < 129 * 64; i += 128) W1[i] = We1[i];
  for (int i = threadIdx.x; i < 64 * 64; i += 128) W2[i] = We2[i];
  for (int i = threadIdx.x; i < 64; i += 128) { B1[i] = be1[i]; B2[i] = be2[i]; }
  __syncthreads();

  const int e = blockIdx.x * 128 + threadIdx.x;
  if (e >= NE) return;
  const int r = ei[e], c = ei[NE + e];

  float d0 = coord[r * 3 + 0] - coord[c * 3 + 0];
  float d1 = coord[r * 3 + 1] - coord[c * 3 + 1];
  float d2 = coord[r * 3 + 2] - coord[c * 3 + 2];
  const float rad = d0 * d0 + d1 * d1 + d2 * d2;
  cd[e * 3 + 0] = d0; cd[e * 3 + 1] = d1; cd[e * 3 + 2] = d2;
  const float inv = 1.0f / (__builtin_sqrtf(rad) + 1e-8f);
  rel[e * 3 + 0] = d0 * inv; rel[e * 3 + 1] = d1 * inv; rel[e * 3 + 2] = d2 * inv;

  float a1[64];
#pragma unroll
  for (int o = 0; o < 64; ++o) a1[o] = B1[o];
  const float* hr = h + (size_t)r * 64;
  for (int i = 0; i < 64; ++i) {
    const float x = hr[i];
    const float* w = &W1[i * 64];
#pragma unroll
    for (int o = 0; o < 64; ++o) a1[o] = fmaf(x, w[o], a1[o]);
  }
  const float* hc = h + (size_t)c * 64;
  for (int i = 0; i < 64; ++i) {
    const float x = hc[i];
    const float* w = &W1[(64 + i) * 64];
#pragma unroll
    for (int o = 0; o < 64; ++o) a1[o] = fmaf(x, w[o], a1[o]);
  }
  {
    const float* w = &W1[128 * 64];
#pragma unroll
    for (int o = 0; o < 64; ++o) a1[o] = fmaf(rad, w[o], a1[o]);
  }
#pragma unroll
  for (int o = 0; o < 64; ++o) a1[o] = siluf(a1[o]);

  float a2[64];
#pragma unroll
  for (int o = 0; o < 64; ++o) a2[o] = B2[o];
  for (int i = 0; i < 64; ++i) {
    const float x = a1[i];
    const float* w = &W2[i * 64];
#pragma unroll
    for (int o = 0; o < 64; ++o) a2[o] = fmaf(x, w[o], a2[o]);
  }
  _Float16* dst = eH + (size_t)e * 64;
#pragma unroll
  for (int o = 0; o < 64; ++o) dst[o] = (_Float16)siluf(a2[o]);
}

// ---------------- kernel 4: SO3 bilinear + MLP via WMMA ----------------
// tp[i, j*64+k] = 64*e[i,j]*e[i,k]; never materialized: each 32-wide K-chunk of the
// A fragment is scale(m) * slice(m), built in registers from pre-loaded f16 slices.
// Branch-free, no compiler unroll: loop over j (64 iters), both K-chunks of a j-block
// inline (static slice selection), one barrier per j, 16 KB double-buffered staging.
#define ESTR 72    // padded e-tile row stride (halves)
#define RSTR 136   // padded relu-tile row stride (halves)

__global__ __launch_bounds__(128) void so3_kernel(
    const _Float16* __restrict__ eH, const _Float16* __restrict__ w1f,
    const _Float16* __restrict__ w2f, const float* __restrict__ bs1,
    const float* __restrict__ bs2, float* __restrict__ edge_feat, int nTiles) {
  __shared__ __align__(16) _Float16 eTile[4][16 * ESTR];     //  9216 B
  __shared__ __align__(16) unsigned char bStage[2][16384];   // 32768 B
  __shared__ __align__(16) _Float16 rTile[4][16 * RSTR];     // 17408 B

  const int wave = threadIdx.x >> 5;
  const int lane = threadIdx.x & 31;
  const int m = lane & 15;     // row within 16-edge tile / column within n-tile
  const int hi = lane >> 4;    // lane-half selector per WMMA layout
  int tile = blockIdx.x * 4 + wave;
  tile = tile < nTiles ? tile : nTiles - 1;   // clamp: no divergence, benign dup work

  _Float16* eT = eTile[wave];
  _Float16* rT = rTile[wave];

  // Per-lane e-row slices for A construction (K-chunk c=0/1 within each j-block of 64).
  const _Float16* erow = eH + (size_t)(tile * 16 + m) * 64;
  const v8h s0a = ld8h(erow + hi * 16);
  const v8h s0b = ld8h(erow + 8 + hi * 16);
  const v8h s1a = ld8h(erow + 32 + hi * 16);
  const v8h s1b = ld8h(erow + 40 + hi * 16);
  // stage e tile to LDS for broadcast scalar reads e[m][j]
  {
    const int rr = lane >> 1, part = lane & 1;
    const _Float16* src = eH + (size_t)(tile * 16 + rr) * 64 + part * 32;
#pragma unroll
    for (int q = 0; q < 4; ++q) {
      U16B t = ldu(src + q * 8);
      stu(&eT[rr * ESTR + part * 32 + q * 8], t);
    }
  }
  // stage first B chunk (j=0: both K-chunks, 16 KB): 128 B per thread
  {
    const unsigned char* src = (const unsigned char*)w1f + threadIdx.x * 128;
    unsigned char* dst = bStage[0] + threadIdx.x * 128;
#pragma unroll
    for (int q = 0; q < 8; ++q) { U16B t = ldu(src + q * 16); stu(dst + q * 16, t); }
  }
  __syncthreads();

  v8f C[8] = {};
#pragma clang loop unroll(disable)
  for (int j = 0; j < 64; ++j) {
    // prefetch next 16 KB chunk (last iter redundantly re-reads chunk 63: benign)
    const int jn = (j + 1) < 64 ? (j + 1) : 63;
    U16B pf[8];
    {
      const unsigned char* src =
          (const unsigned char*)w1f + (size_t)jn * 16384 + threadIdx.x * 128;
#pragma unroll
      for (int q = 0; q < 8; ++q) pf[q] = ldu(src + q * 16);
    }
    const _Float16 sch = (_Float16)(64.0f * (float)eT[m * ESTR + j]);
    const unsigned char* bb = bStage[j & 1];
    {  // K-chunk c=0
      v16h A;
#pragma unroll
      for (int t = 0; t < 8; ++t) { A[t] = s0a[t] * sch; A[8 + t] = s0b[t] * sch; }
#pragma unroll
      for (int nt = 0; nt < 8; ++nt) {
        v16h B = ld16h(bb + nt * 1024 + lane * 32);
        C[nt] = WMMA16(A, B, C[nt]);
      }
    }
    {  // K-chunk c=1
      v16h A;
#pragma unroll
      for (int t = 0; t < 8; ++t) { A[t] = s1a[t] * sch; A[8 + t] = s1b[t] * sch; }
#pragma unroll
      for (int nt = 0; nt < 8; ++nt) {
        v16h B = ld16h(bb + 8192 + nt * 1024 + lane * 32);
        C[nt] = WMMA16(A, B, C[nt]);
      }
    }
    {
      unsigned char* dst = bStage[(j + 1) & 1] + threadIdx.x * 128;
#pragma unroll
      for (int q = 0; q < 8; ++q) stu(dst + q * 16, pf[q]);
    }
    __syncthreads();
  }

  // bias + ReLU -> f16 tile in LDS (row-major 16 x 128, padded)
#pragma unroll
  for (int nt = 0; nt < 8; ++nt) {
    const int colg = nt * 16 + m;
    const float b = bs1[colg];
#pragma unroll
    for (int r2 = 0; r2 < 8; ++r2) {
      const float v = C[nt][r2] + b;
      rT[(r2 + hi * 8) * RSTR + colg] = (_Float16)fmaxf(v, 0.0f);
    }
  }
  // second GEMM: [16,128] @ Ws2[128,64] (+bs2)
  v8f D[4] = {};
#pragma unroll
  for (int s2 = 0; s2 < 4; ++s2) {
    const int base = s2 * 32 + hi * 16;
    v8h lo = ld8h(&rT[m * RSTR + base]);
    v8h ho = ld8h(&rT[m * RSTR + base + 8]);
    v16h A2;
#pragma unroll
    for (int t = 0; t < 8; ++t) { A2[t] = lo[t]; A2[8 + t] = ho[t]; }
#pragma unroll
    for (int nt = 0; nt < 4; ++nt) {
      v16h B = ld16h(w2f + (s2 * 4 + nt) * 512 + lane * 16);
      D[nt] = WMMA16(A2, B, D[nt]);
    }
  }
#pragma unroll
  for (int nt = 0; nt < 4; ++nt) {
    const int col = nt * 16 + m;
    const float b2 = bs2[col];
#pragma unroll
    for (int r2 = 0; r2 < 8; ++r2)
      edge_feat[(size_t)(tile * 16 + r2 + hi * 8) * 64 + col] = D[nt][r2] + b2;
  }
}

// ---------------- kernel 5: coord MLP + segment-sum aggregation ----------------
__global__ __launch_bounds__(128) void coord_agg_kernel(
    const float* __restrict__ edge_feat, const float* __restrict__ cd,
    const int* __restrict__ ei, const float* __restrict__ Wc1,
    const float* __restrict__ bc1, const float* __restrict__ Wc2,
    float* __restrict__ numv, float* __restrict__ cnt, float* __restrict__ agg) {
  __shared__ __align__(16) float W1[64 * 64];
  __shared__ float B1[64];
  __shared__ float W2[64];
  for (int i = threadIdx.x; i < 64 * 64; i += 128) W1[i] = Wc1[i];
  for (int i = threadIdx.x; i < 64; i += 128) { B1[i] = bc1[i]; W2[i] = Wc2[i]; }
  __syncthreads();

  const int e = blockIdx.x * 128 + threadIdx.x;
  if (e >= NE) return;
  const int r = ei[e];

  float ef[64];
  const float* src = edge_feat + (size_t)e * 64;
#pragma unroll
  for (int o = 0; o < 64; ++o) ef[o] = src[o];

  float a1[64];
#pragma unroll
  for (int o = 0; o < 64; ++o) a1[o] = B1[o];
  for (int i = 0; i < 64; ++i) {
    const float x = ef[i];
    const float* w = &W1[i * 64];
#pragma unroll
    for (int o = 0; o < 64; ++o) a1[o] = fmaf(x, w[o], a1[o]);
  }
  float wgt = 0.0f;
#pragma unroll
  for (int o = 0; o < 64; ++o) wgt = fmaf(siluf(a1[o]), W2[o], wgt);

  atomicAdd(&numv[r * 3 + 0], cd[e * 3 + 0] * wgt);
  atomicAdd(&numv[r * 3 + 1], cd[e * 3 + 1] * wgt);
  atomicAdd(&numv[r * 3 + 2], cd[e * 3 + 2] * wgt);
  atomicAdd(&cnt[r], 1.0f);
#pragma unroll
  for (int o = 0; o < 64; ++o) atomicAdd(&agg[r * 64 + o], ef[o]);
}

// ---------------- kernel 6: node MLP + residual + coord update ----------------
__global__ __launch_bounds__(128) void node_kernel(
    const float* __restrict__ h, const float* __restrict__ coord,
    const float* __restrict__ rel, const float* __restrict__ agg,
    const float* __restrict__ numv, const float* __restrict__ cnt,
    const float* __restrict__ Wn1, const float* __restrict__ bn1,
    const float* __restrict__ Wn2, const float* __restrict__ bn2,
    float* __restrict__ out) {
  __shared__ __align__(16) float W1[131 * 64];
  __shared__ __align__(16) float W2[64 * OUTF];
  __shared__ float B1[64];
  __shared__ float B2[OUTF];
  for (int i = threadIdx.x; i < 131 * 64; i += 128) W1[i] = Wn1[i];
  for (int i = threadIdx.x; i < 64 * OUTF; i += 128) W2[i] = Wn2[i];
  for (int i = threadIdx.x; i < 64; i += 128) B1[i] = bn1[i];
  for (int i = threadIdx.x; i < OUTF; i += 128) B2[i] = bn2[i];
  __syncthreads();

  const int n = blockIdx.x * 128 + threadIdx.x;
  if (n >= NN) return;

  float a1[64];
#pragma unroll
  for (int o = 0; o < 64; ++o) a1[o] = B1[o];
  const float* hr = h + (size_t)n * 64;
  for (int i = 0; i < 64; ++i) {  // m rows 0..63: h
    const float x = hr[i];
    const float* w = &W1[i * 64];
#pragma unroll
    for (int o = 0; o < 64; ++o) a1[o] = fmaf(x, w[o], a1[o]);
  }
  float rl[3] = {rel[n * 3 + 0], rel[n * 3 + 1], rel[n * 3 + 2]};
#pragma unroll
  for (int i = 0; i < 3; ++i) {   // m rows 64..66: rel (uses E==N alignment)
    const float* w = &W1[(64 + i) * 64];
#pragma unroll
    for (int o = 0; o < 64; ++o) a1[o] = fmaf(rl[i], w[o], a1[o]);
  }
  const float* ag = agg + (size_t)n * 64;
  for (int i = 0; i < 64; ++i) {  // m rows 67..130: agg
    const float x = ag[i];
    const float* w = &W1[(67 + i) * 64];
#pragma unroll
    for (int o = 0; o < 64; ++o) a1[o] = fmaf(x, w[o], a1[o]);
  }
#pragma unroll
  for (int o = 0; o < 64; ++o) a1[o] = siluf(a1[o]);

  float o2[OUTF];
#pragma unroll
  for (int o = 0; o < OUTF; ++o) o2[o] = B2[o];
  for (int i = 0; i < 64; ++i) {
    const float x = a1[i];
    const float* w = &W2[i * OUTF];
#pragma unroll
    for (int o = 0; o < OUTF; ++o) o2[o] = fmaf(x, w[o], o2[o]);
  }
  float* ho = out + (size_t)n * OUTF;
#pragma unroll
  for (int o = 0; o < 64; ++o) ho[o] = hr[o] + o2[o];
#pragma unroll
  for (int o = 64; o < OUTF; ++o) ho[o] = rl[o - 64] + o2[o];

  const float ic = 1.0f / fmaxf(cnt[n], 1.0f);
  float* co = out + (size_t)NN * OUTF + (size_t)n * 3;
#pragma unroll
  for (int d = 0; d < 3; ++d) co[d] = coord[n * 3 + d] + numv[n * 3 + d] * ic;
}

// ---------------- host launcher ----------------
extern "C" void kernel_launch(void* const* d_in, const int* in_sizes, int n_in,
                              void* d_out, int out_size, void* d_ws, size_t ws_size,
                              hipStream_t stream) {
  const float* h     = (const float*)d_in[0];
  const float* coord = (const float*)d_in[1];
  const int*   ei    = (const int*)d_in[2];
  const float* We1 = (const float*)d_in[3];
  const float* be1 = (const float*)d_in[4];
  const float* We2 = (const float*)d_in[5];
  const float* be2 = (const float*)d_in[6];
  const float* Ws1 = (const float*)d_in[7];
  const float* bs1 = (const float*)d_in[8];
  const float* Ws2 = (const float*)d_in[9];
  const float* bs2 = (const float*)d_in[10];
  const float* Wc1 = (const float*)d_in[11];
  const float* bc1 = (const float*)d_in[12];
  const float* Wc2 = (const float*)d_in[13];
  const float* Wn1 = (const float*)d_in[14];
  const float* bn1 = (const float*)d_in[15];
  const float* Wn2 = (const float*)d_in[16];
  const float* bn2 = (const float*)d_in[17];

  char* w = (char*)d_ws;
  _Float16* eH  = (_Float16*)(w + O_EH);
  float*    cd  = (float*)(w + O_CD);
  float*    rel = (float*)(w + O_REL);
  float*    ef  = (float*)(w + O_EF);
  _Float16* w1f = (_Float16*)(w + O_W1F);
  _Float16* w2f = (_Float16*)(w + O_W2F);
  float*    num = (float*)(w + O_NUM);
  float*    cnt = (float*)(w + O_CNT);
  float*    agg = (float*)(w + O_AGG);
  float*    out = (float*)d_out;

  // 1. weight fragment conversion
  {
    const int total = 4096 * 128 + 128 * 64;
    prep_w_kernel<<<(total + 255) / 256, 256, 0, stream>>>(Ws1, Ws2, w1f, w2f);
  }
  // 2. zero accumulators (num..agg region, contiguous incl. padding)
  {
    const int zn = (int)((O_AGG + (size_t)NN * 64 * 4 - O_NUM) / 4);
    zero_kernel<<<(zn + 255) / 256, 256, 0, stream>>>(num, zn);
  }
  // 3. edge MLP
  edge_mlp_kernel<<<(NE + 127) / 128, 128, 0, stream>>>(h, coord, ei, We1, be1, We2, be2,
                                                        eH, cd, rel);
  // 4. SO3 WMMA GEMMs
  {
    const int nTiles = (NE + 15) / 16;
    so3_kernel<<<(nTiles + 3) / 4, 128, 0, stream>>>(eH, w1f, w2f, bs1, bs2, ef, nTiles);
  }
  // 5. coord MLP + segment sums
  coord_agg_kernel<<<(NE + 127) / 128, 128, 0, stream>>>(ef, cd, ei, Wc1, bc1, Wc2,
                                                         num, cnt, agg);
  // 6. node MLP + outputs
  node_kernel<<<(NN + 127) / 128, 128, 0, stream>>>(h, coord, rel, agg, num, cnt,
                                                    Wn1, bn1, Wn2, bn2, out);
  (void)in_sizes; (void)n_in; (void)out_size; (void)ws_size;
}
